// Decoder_31576599561086
// MI455X (gfx1250) — compile-verified, matrix operand order checked by reading, not run on hardware
//
#include <hip/hip_runtime.h>
#include <hip/hip_bf16.h>
#include <math.h>
#include <stdint.h>

#define B_DIM 64
#define L_DIM 32
#define T_DIM 2048
#define V_DIM 32000
#define E_DIM 128
#define H_DIM 128
#define SOS_ID 0
#define NEG_INF_F (-1e30f)

typedef __attribute__((ext_vector_type(16))) _Float16 v16h;
typedef __attribute__((ext_vector_type(8)))  _Float16 v8h;
typedef __attribute__((ext_vector_type(8)))  float    v8f;
typedef unsigned int u32x4v __attribute__((ext_vector_type(4)));
typedef int          i32x8v __attribute__((ext_vector_type(8)));
typedef int          i32x4v __attribute__((ext_vector_type(4)));

// ---------------------------------------------------------------------------
// Fragment helpers. Per CDNA5 ISA 7.12.2, a 16-bit A/B fragment's 16 K-values
// per lane are two contiguous runs of 8: [k0+half*8 .. +7] and [+16 .. +23].
// ---------------------------------------------------------------------------
__device__ inline v16h frag_from_runs(v8h lo, v8h hi) {  // f16 source
  v16h f;
#pragma unroll
  for (int i = 0; i < 8; ++i) { f[i] = lo[i]; f[8 + i] = hi[i]; }
  return f;
}

__device__ inline v16h frag_from_f32_runs(const float* __restrict__ pa) {
  // Two 32B runs -> 4x b128 loads + pack-convert to f16.
  const float4 r0 = ((const float4*)pa)[0];
  const float4 r1 = ((const float4*)pa)[1];
  const float4 r2 = ((const float4*)pa)[4];   // +16 floats
  const float4 r3 = ((const float4*)pa)[5];
  v16h f;
  f[0]  = (_Float16)r0.x; f[1]  = (_Float16)r0.y;
  f[2]  = (_Float16)r0.z; f[3]  = (_Float16)r0.w;
  f[4]  = (_Float16)r1.x; f[5]  = (_Float16)r1.y;
  f[6]  = (_Float16)r1.z; f[7]  = (_Float16)r1.w;
  f[8]  = (_Float16)r2.x; f[9]  = (_Float16)r2.y;
  f[10] = (_Float16)r2.z; f[11] = (_Float16)r2.w;
  f[12] = (_Float16)r3.x; f[13] = (_Float16)r3.y;
  f[14] = (_Float16)r3.z; f[15] = (_Float16)r3.w;
  return f;
}

// ---------------------------------------------------------------------------
// TDM: stage a [128 rows x 128 k] f16 panel (32KB) from global into LDS.
// 2D D# descriptor: data_size=2B, tile 128(contig) x 128, stride 128.
// Call from a single wave; caller must __syncthreads() afterwards.
// ---------------------------------------------------------------------------
__device__ inline void tdm_stage_panel(const _Float16* __restrict__ gptr,
                                       _Float16* lds_panel,
                                       unsigned tensor_rows) {
#if __has_builtin(__builtin_amdgcn_tensor_load_to_lds)
  const unsigned long long ga = (unsigned long long)(uintptr_t)gptr;
  u32x4v g0;
  g0[0] = 1u;                                      // count=1, user mode
  g0[1] = (unsigned)(uintptr_t)lds_panel;          // lds_addr (byte offset)
  g0[2] = (unsigned)(ga & 0xFFFFFFFFu);            // global_addr[31:0]
  g0[3] = (unsigned)((ga >> 32) & 0x1FFFFFFu) | (2u << 30);   // type=2
  i32x8v g1;
  g1[0] = 0x00010000;                 // data_size=1 (2 bytes)
  g1[1] = (int)(128u << 16);          // tensor_dim0=128 (low16 @ [31:16])
  g1[2] = (int)(tensor_rows << 16);   // dim0 hi=0 | tensor_dim1 low16
  g1[3] = (int)(128u << 16);          // dim1 hi=0 | tile_dim0=128
  g1[4] = 128;                        // tile_dim1=128, tile_dim2=0
  g1[5] = 128;                        // tensor_dim0_stride=128 elements
  g1[6] = 0;
  g1[7] = 0;
  i32x4v gz4 = {0, 0, 0, 0};
  i32x8v gz8 = {0, 0, 0, 0, 0, 0, 0, 0};
  __builtin_amdgcn_tensor_load_to_lds(g0, g1, gz4, gz4, gz8, 0);
  __builtin_amdgcn_s_wait_tensorcnt(0);
#else
  for (int i = threadIdx.x & 31; i < 128 * 128 / 8; i += 32)
    ((v8h*)lds_panel)[i] = ((const v8h*)gptr)[i];
#endif
}

// ---------------------------------------------------------------------------
// Transpose+convert W (K=128 x N, f32 row-major) -> WT_h (N x 128, f16).
// Used for fc_k (N=32000) and encW_k (N=128). 32x32 tiles through LDS.
// ---------------------------------------------------------------------------
__global__ __launch_bounds__(256) void transpose_convert_kernel(
    const float* __restrict__ src, _Float16* __restrict__ dst, int N) {
  __shared__ float tile[32][33];
  const int bk = (blockIdx.x & 3) * 32;     // K block (128/32 = 4)
  const int bn = (blockIdx.x >> 2) * 32;    // N block
  const int tx = threadIdx.x & 31;
  const int ty = threadIdx.x >> 5;          // 0..7
#pragma unroll
  for (int i = 0; i < 4; ++i)
    tile[ty + i * 8][tx] = src[(size_t)(bk + ty + i * 8) * N + bn + tx];
  __syncthreads();
#pragma unroll
  for (int i = 0; i < 4; ++i)
    dst[(size_t)(bn + ty + i * 8) * H_DIM + bk + tx] =
        (_Float16)tile[tx][ty + i * 8];
}

// ---------------------------------------------------------------------------
// enc_feat GEMM: (131072 x 128) f32 @ encW (128x128) + bias.
// One 128-col strip total; B panel TDM-staged to LDS, B-fragments
// loop-invariant; each block sweeps 16 row tiles (256 rows).
// ---------------------------------------------------------------------------
__global__ __launch_bounds__(256) void enc_gemm_kernel(
    const float* __restrict__ A,       // (B*T, 128) f32
    const _Float16* __restrict__ BTh,  // (128, 128) f16 = encW_k^T
    const float* __restrict__ bias, float* __restrict__ C) {
  __shared__ __attribute__((aligned(16))) _Float16 lb[128 * 128];
  const int tid  = threadIdx.x;
  const int lane = tid & 31;
  const int wave = tid >> 5;
  const int mpart = blockIdx.x;               // 0..511

  if (wave == 0) tdm_stage_panel(BTh, lb, 128u);
  __syncthreads();

  const int nloc = wave * 16 + (lane & 15);
  const int half = lane >> 4;
  const int m    = lane & 15;

  v16h bf[4];
#pragma unroll
  for (int c = 0; c < 4; ++c) {
    const _Float16* p = lb + nloc * H_DIM + c * 32 + half * 8;
    bf[c] = frag_from_runs(*(const v8h*)p, *(const v8h*)(p + 16));
  }
  const float bv = bias[nloc];

  for (int tm = mpart * 16; tm < mpart * 16 + 16; ++tm) {
    v8f acc = {};
#pragma unroll
    for (int c = 0; c < 4; ++c) {
      const float* pa = A + (size_t)(tm * 16 + m) * H_DIM + c * 32 + half * 8;
      v16h af = frag_from_f32_runs(pa);
      acc = __builtin_amdgcn_wmma_f32_16x16x32_f16(
          false, af, false, bf[c], (short)0, acc, false, false);
    }
    const int row0 = tm * 16 + (half << 3);
#pragma unroll
    for (int g = 0; g < 8; ++g)
      C[(size_t)(row0 + g) * H_DIM + nloc] = acc[g] + bv;
  }
}

// ---------------------------------------------------------------------------
// fc GEMM: logits(2048x32000) = ctx_h(2048x128,f16) @ fcT_h^T + fc_b.
// Block = 128-column strip; B panel TDM-staged; B-fragments loop-invariant;
// each block sweeps 16 row tiles.
// ---------------------------------------------------------------------------
__global__ __launch_bounds__(256) void fc_gemm_kernel(
    const _Float16* __restrict__ Ah,   // (2048, 128) f16 row-major
    const _Float16* __restrict__ BTh,  // (32000, 128) f16 (fc_k transposed)
    const float* __restrict__ bias, float* __restrict__ C) {
  __shared__ __attribute__((aligned(16))) _Float16 lb[128 * 128];
  const int tid  = threadIdx.x;
  const int lane = tid & 31;
  const int wave = tid >> 5;
  const int strip = blockIdx.x % (V_DIM / 128);   // 0..249
  const int mpart = blockIdx.x / (V_DIM / 128);   // 0..7
  const int n0 = strip * 128;

  if (wave == 0) tdm_stage_panel(BTh + (size_t)n0 * H_DIM, lb, (unsigned)V_DIM);
  __syncthreads();

  const int nloc = wave * 16 + (lane & 15);
  const int half = lane >> 4;
  const int m    = lane & 15;

  v16h bf[4];
#pragma unroll
  for (int c = 0; c < 4; ++c) {
    const _Float16* p = lb + nloc * H_DIM + c * 32 + half * 8;
    bf[c] = frag_from_runs(*(const v8h*)p, *(const v8h*)(p + 16));
  }
  const int gn = n0 + nloc;
  const float bv = bias[gn];

  for (int tm = mpart * 16; tm < mpart * 16 + 16; ++tm) {
    v8f acc = {};
#pragma unroll
    for (int c = 0; c < 4; ++c) {
      const _Float16* pa =
          Ah + (size_t)(tm * 16 + m) * H_DIM + c * 32 + half * 8;
      v16h af = frag_from_runs(*(const v8h*)pa, *(const v8h*)(pa + 16));
      acc = __builtin_amdgcn_wmma_f32_16x16x32_f16(
          false, af, false, bf[c], (short)0, acc, false, false);
    }
    const int row0 = tm * 16 + (half << 3);
#pragma unroll
    for (int g = 0; g < 8; ++g)
      C[(size_t)(row0 + g) * V_DIM + gn] = acc[g] + bv;
  }
}

// ---------------------------------------------------------------------------
// GRU + attention recurrence. One block (256 threads) per batch element,
// 32 sequential steps; h/scores in LDS; ctx emitted as f16 for the fc GEMM.
// ---------------------------------------------------------------------------
__global__ __launch_bounds__(256) void gru_attn_kernel(
    const int* __restrict__ target_input,
    const unsigned char* __restrict__ weight_mask,
    const float* __restrict__ embed,
    const float* __restrict__ gate_k, const float* __restrict__ gate_b,
    const float* __restrict__ cand_k, const float* __restrict__ cand_b,
    const float* __restrict__ decW_k, const float* __restrict__ decW_b,
    const float* __restrict__ V_k, const float* __restrict__ V_b,
    const float* __restrict__ enc_feat,
    _Float16* __restrict__ ctx_out, float* __restrict__ attn_out) {
  __shared__ float sc[T_DIM];
  __shared__ float xs[H_DIM], hs[H_DIM], hns[H_DIM];
  __shared__ float rs[H_DIM], us[H_DIM], wv[H_DIM];
  __shared__ float red[256];

  const int b = blockIdx.x;
  const int tid = threadIdx.x;
  const float* __restrict__ ef = enc_feat + (size_t)b * T_DIM * H_DIM;
  const unsigned char* __restrict__ mk = weight_mask + (size_t)b * T_DIM;
  const float vb = V_b[0];

  if (tid < H_DIM) hs[tid] = 0.0f;
  __syncthreads();

  for (int l = 0; l < L_DIM; ++l) {
    const int tok = (l == 0) ? SOS_ID : target_input[b * L_DIM + l - 1];
    if (tid < H_DIM) xs[tid] = embed[(size_t)tok * E_DIM + tid];
    __syncthreads();

    {  // gates = sigmoid([x,h] @ gate_k + gate_b)
      float acc = gate_b[tid];
#pragma unroll 4
      for (int j = 0; j < E_DIM + H_DIM; ++j) {
        const float xv = (j < E_DIM) ? xs[j] : hs[j - E_DIM];
        acc += xv * gate_k[(size_t)j * (2 * H_DIM) + tid];
      }
      const float g = 1.0f / (1.0f + expf(-acc));
      if (tid < H_DIM) rs[tid] = g;
      else             us[tid - H_DIM] = g;
    }
    __syncthreads();

    if (tid < H_DIM) {  // c = tanh([x, r*h] @ cand_k + b); h_new
      float acc = cand_b[tid];
#pragma unroll 4
      for (int j = 0; j < E_DIM + H_DIM; ++j) {
        const float xv = (j < E_DIM) ? xs[j] : rs[j - E_DIM] * hs[j - E_DIM];
        acc += xv * cand_k[(size_t)j * H_DIM + tid];
      }
      const float c = tanhf(acc);
      const float u = us[tid];
      hns[tid] = u * hs[tid] + (1.0f - u) * c;
    }
    __syncthreads();

    if (tid < H_DIM) {  // wv = (h_new @ decW_k + b) * V_k
      float acc = decW_b[tid];
#pragma unroll 4
      for (int j = 0; j < H_DIM; ++j)
        acc += hns[j] * decW_k[(size_t)j * H_DIM + tid];
      wv[tid] = acc * V_k[tid];
    }
    __syncthreads();

    float lmax = NEG_INF_F;
    for (int t = tid; t < T_DIM; t += 256) {
      const float4* __restrict__ row = (const float4*)(ef + (size_t)t * H_DIM);
      float s = 0.0f;
#pragma unroll 8
      for (int j4 = 0; j4 < H_DIM / 4; ++j4) {
        const float4 e = row[j4];
        s += wv[4 * j4 + 0] * e.x + wv[4 * j4 + 1] * e.y +
             wv[4 * j4 + 2] * e.z + wv[4 * j4 + 3] * e.w;
      }
      s += vb;
      if (!mk[t]) s = NEG_INF_F;
      sc[t] = s;
      lmax = fmaxf(lmax, s);
    }
    red[tid] = lmax;
    __syncthreads();
    for (int st = 128; st > 0; st >>= 1) {
      if (tid < st) red[tid] = fmaxf(red[tid], red[tid + st]);
      __syncthreads();
    }
    const float mx = red[0];
    __syncthreads();

    float lsum = 0.0f;
    for (int t = tid; t < T_DIM; t += 256) {
      const float e = expf(sc[t] - mx);
      sc[t] = e;
      lsum += e;
    }
    red[tid] = lsum;
    __syncthreads();
    for (int st = 128; st > 0; st >>= 1) {
      if (tid < st) red[tid] += red[tid + st];
      __syncthreads();
    }
    const float inv = 1.0f / red[0];
    __syncthreads();

    float* __restrict__ arow = attn_out + ((size_t)b * L_DIM + l) * T_DIM;
    for (int t = tid; t < T_DIM; t += 256) {
      const float a = sc[t] * inv;
      sc[t] = a;
      arow[t] = a;
    }
    __syncthreads();

    {  // ctx[o] = sum_t attn[t] * enc_feat[t,o]
      const int o = tid & (H_DIM - 1);
      const int half = tid >> 7;
      const int t0 = half * (T_DIM / 2);
      float acc = 0.0f;
      for (int t = t0; t < t0 + T_DIM / 2; ++t)
        acc += sc[t] * ef[(size_t)t * H_DIM + o];
      red[tid] = acc;
    }
    __syncthreads();
    if (tid < H_DIM) {
      ctx_out[((size_t)b * L_DIM + l) * H_DIM + tid] =
          (_Float16)(red[tid] + red[tid + H_DIM]);
      hs[tid] = hns[tid];
    }
    __syncthreads();
  }
}

// ---------------------------------------------------------------------------
extern "C" void kernel_launch(void* const* d_in, const int* in_sizes, int n_in,
                              void* d_out, int out_size, void* d_ws, size_t ws_size,
                              hipStream_t stream) {
  const int*   target_input = (const int*)d_in[0];
  const float* enc_output   = (const float*)d_in[1];
  const unsigned char* weight_mask = (const unsigned char*)d_in[2];
  const float* embed  = (const float*)d_in[3];
  const float* gate_k = (const float*)d_in[4];
  const float* gate_b = (const float*)d_in[5];
  const float* cand_k = (const float*)d_in[6];
  const float* cand_b = (const float*)d_in[7];
  const float* encW_k = (const float*)d_in[8];
  const float* encW_b = (const float*)d_in[9];
  const float* decW_k = (const float*)d_in[10];
  const float* decW_b = (const float*)d_in[11];
  const float* V_k    = (const float*)d_in[12];
  const float* V_b    = (const float*)d_in[13];
  const float* fc_k   = (const float*)d_in[14];
  const float* fc_b   = (const float*)d_in[15];

  float* logits = (float*)d_out;                            // (B,L,V)
  float* attns  = logits + (size_t)B_DIM * L_DIM * V_DIM;   // (B,L,T)

  char* ws = (char*)d_ws;
  size_t off = 0;
  float*    enc_feat = (float*)(ws + off);      off += (size_t)B_DIM * T_DIM * H_DIM * 4;
  _Float16* fcT_h    = (_Float16*)(ws + off);   off += (size_t)V_DIM * H_DIM * 2;
  _Float16* ctx_h    = (_Float16*)(ws + off);   off += (size_t)B_DIM * L_DIM * H_DIM * 2;
  _Float16* encWT_h  = (_Float16*)(ws + off);   off += (size_t)H_DIM * H_DIM * 2;

  // 0) weight panels -> transposed f16 (independent of everything else)
  transpose_convert_kernel<<<4 * (V_DIM / 32), 256, 0, stream>>>(fc_k, fcT_h, V_DIM);
  transpose_convert_kernel<<<4 * (H_DIM / 32), 256, 0, stream>>>(encW_k, encWT_h, H_DIM);

  // 1) enc_feat = enc_output @ encW_k + encW_b   (131072 x 128 x 128)
  enc_gemm_kernel<<<(B_DIM * T_DIM) / 256, 256, 0, stream>>>(
      enc_output, encWT_h, encW_b, enc_feat);

  // 2) sequential GRU + attention; emits attn (output) and ctx_h (workspace)
  gru_attn_kernel<<<B_DIM, 256, 0, stream>>>(
      target_input, weight_mask, embed, gate_k, gate_b, cand_k, cand_b,
      decW_k, decW_b, V_k, V_b, enc_feat, ctx_h, attns);

  // 3) logits = ctx @ fc_k + fc_b   (2048 x 32000 x 128), TDM-staged B panels
  fc_gemm_kernel<<<(V_DIM / 128) * 8, 256, 0, stream>>>(
      ctx_h, fcT_h, fc_b, logits);
}
